// ChunkedSelfAttention_90142773609261
// MI455X (gfx1250) — matrix operand order repaired and checked by reading, not compiled, not run on hardware
//
#include <hip/hip_runtime.h>
#include <math.h>

typedef __attribute__((ext_vector_type(16))) _Float16 v16h;
typedef __attribute__((ext_vector_type(8)))  float    v8f;

#define NB 4
#define NL 4096
#define NH 16
#define ND 128
#define NCHUNK 1024

// A-matrix fragment (16x32 f16): lane L holds row M=L%16; element e -> VGPR e/2.
// K(e) = 2*(r&3) + p + 8*half + 16*(r>>2), r=e/2, p=e&1, half=L/16.
__device__ __forceinline__ v16h ldsA(const _Float16* base, int stride, int lane) {
  const int M = lane & 15, half = lane >> 4;
  v16h a;
#pragma unroll
  for (int e = 0; e < 16; ++e) {
    const int r = e >> 1, p = e & 1;
    const int K = ((r & 3) << 1) + p + (half << 3) + ((r >> 2) << 4);
    a[e] = base[M * stride + K];
  }
  return a;
}

// B-matrix fragment (32x16 f16): lane L holds column N=L%16; K(e) = (e&7) + 8*half + 16*(e>>3).
// base is row-major [N][k] (i.e. already "transposed" source: B[k][n] = base[n*stride + k]).
__device__ __forceinline__ v16h ldsB(const _Float16* base, int stride, int lane) {
  const int N = lane & 15, half = lane >> 4;
  v16h bv;
#pragma unroll
  for (int e = 0; e < 16; ++e)
    bv[e] = base[N * stride + (half << 3) + (e & 7) + ((e >> 3) << 4)];
  return bv;
}

__global__ __launch_bounds__(256, 1)
void attn_fwd(const float* __restrict__ qg, const float* __restrict__ kg,
              const float* __restrict__ vg, float* __restrict__ out) {
  __shared__ _Float16 Ksc[2][32][130];   // [buf][key][feature] (RoPE'd, f16)
  __shared__ _Float16 Vsc[2][128][34];   // [buf][dv][key]      (transposed, f16)
  __shared__ _Float16 Qsc[8][16][130];   // [wave][row][feature]
  __shared__ _Float16 Psc[8][16][34];    // [wave][row][key-in-block]

  const int tid  = threadIdx.x;
  const int wave = tid >> 5;
  const int lane = tid & 31;
  const int g  = blockIdx.x;
  const int qgrp = g & 7;
  const int h  = (g >> 3) & (NH - 1);
  const int nc = (g >> 7) & 3;
  const int b  = g >> 9;

  const int qt     = qgrp * 8 + wave;      // q tile 0..63 within chunk
  const int nblk   = (qt >> 1) + 1;        // causal 32-key blocks for this wave
  const int maxblk = qgrp * 4 + 4;         // max over the 8 waves
  const int half = lane >> 4;
  const int col  = lane & 15;

  const float ropec  = 0.14391156831f;         // ln(10000)/64
  const float qscale = 0.08838834764831845f;   // 1/sqrt(128)

  // ---- stage this wave's Q tile: RoPE + scale -> f16 LDS ----
  {
    const int posbase = nc * NCHUNK + qt * 16;
    for (int it = 0; it < 32; ++it) {
      const int idx = it * 32 + lane;
      const int rr = idx >> 6;
      const int i  = idx & 63;
      const int pos = posbase + rr;
      const float* p = qg + (((long)(b * NL + pos) * NH + h) << 7);
      const float x1 = p[i], x2 = p[i + 64];
      const float fr = __expf(-(float)i * ropec);
      float sn, cs; __sincosf((float)pos * fr, &sn, &cs);
      Qsc[wave][rr][i]      = (_Float16)((x1 * cs - x2 * sn) * qscale);
      Qsc[wave][rr][i + 64] = (_Float16)((x2 * cs + x1 * sn) * qscale);
    }
  }

  // ---- cooperative K/V block staging helpers (all 256 threads) ----
  float kx[16], vx[16];
  auto kvload = [&](int j) {
    const int r = tid >> 3, c = tid & 7;
    const int pos = nc * NCHUNK + j * 32 + r;
    const long base = ((long)(b * NL + pos) * NH + h) << 7;
#pragma unroll
    for (int gq = 0; gq < 4; ++gq) {
      const float4 kk = *(const float4*)(kg + base + c * 4 + gq * 32);
      const float4 vv = *(const float4*)(vg + base + c * 4 + gq * 32);
      kx[gq * 4 + 0] = kk.x; kx[gq * 4 + 1] = kk.y; kx[gq * 4 + 2] = kk.z; kx[gq * 4 + 3] = kk.w;
      vx[gq * 4 + 0] = vv.x; vx[gq * 4 + 1] = vv.y; vx[gq * 4 + 2] = vv.z; vx[gq * 4 + 3] = vv.w;
    }
  };
  auto kvstore = [&](int j) {
    const int r = tid >> 3, c = tid & 7;
    const int buf = j & 1;
    const int pos = nc * NCHUNK + j * 32 + r;
#pragma unroll
    for (int gq = 0; gq < 2; ++gq)
#pragma unroll
      for (int e = 0; e < 4; ++e) {
        const int i = c * 4 + gq * 32 + e;
        const float x1 = kx[gq * 4 + e];
        const float x2 = kx[(gq + 2) * 4 + e];
        const float fr = __expf(-(float)i * ropec);
        float sn, cs; __sincosf((float)pos * fr, &sn, &cs);
        Ksc[buf][r][i]      = (_Float16)(x1 * cs - x2 * sn);
        Ksc[buf][r][i + 64] = (_Float16)(x2 * cs + x1 * sn);
      }
#pragma unroll
    for (int gq = 0; gq < 4; ++gq)
#pragma unroll
      for (int e = 0; e < 4; ++e)
        Vsc[buf][c * 4 + gq * 32 + e][r] = (_Float16)vx[gq * 4 + e];
  };

  kvload(0);
  kvstore(0);
  __syncthreads();

  // ---- per-wave persistent state ----
  v16h qf[4];
#pragma unroll
  for (int s = 0; s < 4; ++s)
    qf[s] = ldsA(&Qsc[wave][0][s * 32], 130, lane);

  float m[8], l[8];
  v8f acc[8];
#pragma unroll
  for (int r = 0; r < 8; ++r) { m[r] = -INFINITY; l[r] = 0.0f; }
#pragma unroll
  for (int t = 0; t < 8; ++t)
#pragma unroll
    for (int r = 0; r < 8; ++r) acc[t][r] = 0.0f;

  // ---- main loop over 32-key blocks (double buffered) ----
  for (int j = 0; j < maxblk; ++j) {
    const int cur = j & 1;
    const bool havenext = (j + 1 < maxblk);
    if (havenext) kvload(j + 1);   // global loads overlap compute below

    if (j < nblk) {                // wave-uniform predicate: EXEC stays all-ones
      v8f S0, S1;
#pragma unroll
      for (int r = 0; r < 8; ++r) { S0[r] = 0.0f; S1[r] = 0.0f; }
#pragma unroll
      for (int s = 0; s < 4; ++s) {
        const v16h b0 = ldsB(&Ksc[cur][0][s * 32],  130, lane);
        const v16h b1 = ldsB(&Ksc[cur][16][s * 32], 130, lane);
        S0 = __builtin_amdgcn_wmma_f32_16x16x32_f16(false, qf[s], false, b0, (short)0, S0, false, false);
        S1 = __builtin_amdgcn_wmma_f32_16x16x32_f16(false, qf[s], false, b1, (short)0, S1, false, false);
      }

      const int key0 = j * 32 + col;
      const int key1 = key0 + 16;
#pragma unroll
      for (int r = 0; r < 8; ++r) {
        const int row = qt * 16 + half * 8 + r;   // chunk-local q position
        float s0 = (key0 <= row) ? S0[r] : -INFINITY;
        float s1 = (key1 <= row) ? S1[r] : -INFINITY;
        float mx = fmaxf(s0, s1);
#pragma unroll
        for (int msk = 8; msk >= 1; msk >>= 1)
          mx = fmaxf(mx, __shfl_xor(mx, msk, 32));
        const float mn = fmaxf(m[r], mx);
        const float alpha = __expf(m[r] - mn);
        m[r] = mn;
        const float p0 = __expf(s0 - mn);
        const float p1 = __expf(s1 - mn);
        float rs = p0 + p1;
#pragma unroll
        for (int msk = 8; msk >= 1; msk >>= 1)
          rs += __shfl_xor(rs, msk, 32);
        l[r] = l[r] * alpha + rs;
#pragma unroll
        for (int t = 0; t < 8; ++t) acc[t][r] *= alpha;
        Psc[wave][half * 8 + r][col]      = (_Float16)p0;
        Psc[wave][half * 8 + r][col + 16] = (_Float16)p1;
      }
      __builtin_amdgcn_fence(__ATOMIC_ACQ_REL, "workgroup");  // order Psc store->load (intra-wave, cross-lane)

      const v16h pf = ldsA(&Psc[wave][0][0], 34, lane);
#pragma unroll
      for (int t = 0; t < 8; ++t) {
        const v16h bv = ldsB(&Vsc[cur][t * 16][0], 34, lane);
        acc[t] = __builtin_amdgcn_wmma_f32_16x16x32_f16(false, pf, false, bv, (short)0, acc[t], false, false);
      }
    }

    if (havenext) kvstore(j + 1);  // writes buffer (j+1)&1, disjoint from 'cur'
    __syncthreads();
  }

  // ---- epilogue: normalize and store ----
#pragma unroll
  for (int r = 0; r < 8; ++r) {
    const int row = qt * 16 + half * 8 + r;
    const int pos = nc * NCHUNK + row;
    const float invl = 1.0f / l[r];
    float* op = out + (((long)(b * NL + pos) * NH + h) << 7);
#pragma unroll
    for (int t = 0; t < 8; ++t)
      op[t * 16 + col] = acc[t][r] * invl;
  }
}

extern "C" void kernel_launch(void* const* d_in, const int* in_sizes, int n_in,
                              void* d_out, int out_size, void* d_ws, size_t ws_size,
                              hipStream_t stream) {
  (void)in_sizes; (void)n_in; (void)out_size; (void)d_ws; (void)ws_size;
  const float* q = (const float*)d_in[0];
  const float* k = (const float*)d_in[1];
  const float* v = (const float*)d_in[2];
  float* out = (float*)d_out;
  dim3 grid(NB * 4 * NH * 8);   // b * chunks * heads * q-tile-groups = 2048
  dim3 block(256);
  hipLaunchKernelGGL(attn_fwd, grid, block, 0, stream, q, k, v, out);
}
